// RNN_63024350102007
// MI455X (gfx1250) — matrix operand order repaired and checked by reading, not compiled, not run on hardware
//
#include <hip/hip_runtime.h>

// ---------------------------------------------------------------------------
// Elman RNN for MI455X (gfx1250, wave32, WMMA).
//
//   Phase P (prep):  swizzle w0 / w_out (fp32) into f16 WMMA fragment layouts
//                    in workspace (L2-resident, 8.9 MB + 256 KB).
//   Phase R (recur): persistent kernel, 32 WGs x 512 threads (16 waves).
//                    WG owns 64 neurons = 4 M-tiles; 4 waves cooperate per
//                    M-tile, splitting K=2176 (padded) into 4 x 17 WMMA
//                    chunks. The compiler hoists each wave's 17 A-fragments
//                    (136 VGPRs of weights) out of the 8192-step loop, so the
//                    steady state is ds_load_b128 (z) + v_wmma only.
//                    B loads are software-pipelined depth-2; LDS reduce,
//                    fast sigmoid, 1x b128 state store, grid barrier per step.
//   Phase O (GEMM):  out = sigmoid(mem_all @ w_out + b_out), dense
//                    8192x64x2048 WMMA GEMM.
// ---------------------------------------------------------------------------

#define T_STEPS   8192
#define SIZE_IN   64
#define SIZE_MEM  2048
#define SIZE_OUT  64
#define K_TOT     (SIZE_IN + SIZE_MEM)   // 2112
#define NCHUNK_P  68                     // K chunks padded to 4*17 (zero tail)
#define K_PAD     (NCHUNK_P * 32)        // 2176
#define CH_PER_W  (NCHUNK_P / 4)         // 17 chunks per cooperating wave
#define OCHUNK    (SIZE_MEM / 32)        // 64 K-chunks for the output GEMM
#define G_WG      32                     // workgroups in recurrent kernel
#define MT_PER_WG 4                      // 16-neuron M-tiles per WG
#define MT_TOT    (SIZE_MEM / 16)        // 128 global M-tiles over neurons

typedef __attribute__((ext_vector_type(16))) _Float16 v16h;
typedef __attribute__((ext_vector_type(8)))  _Float16 v8h;
typedef __attribute__((ext_vector_type(8)))  float    v8f;

// Fast sigmoid: v_exp_f32 + v_rcp_f32 (no IEEE div chain); result is stored
// as f16, so ~1 ulp rcp error is irrelevant.
__device__ __forceinline__ float fast_sigmoid(float x) {
    return __builtin_amdgcn_rcpf(1.0f + __expf(-x));
}

// K offset of half h (0..15) inside a 32-wide K chunk for the 16-bit
// A-matrix VGPR layout (ISA 7.12.2): lanes 0-15 hold K {0..7,16..23},
// lanes 16-31 hold K {8..15,24..31}.
__device__ __forceinline__ int a_koff(int h, int hi) {
    return ((h < 8) ? h : h + 8) + 8 * hi;
}

// ---------------------------------------------------------------------------
// Prep: w0 (2112x2048 fp32, [k][n]) -> f16 A-fragments, zero-padded to K=2176.
// Fragment address: ((mt*NCHUNK_P + ch)*32 + lane)*16 + h.
// ---------------------------------------------------------------------------
__global__ void k_prep_w0(const float* __restrict__ w0,
                          _Float16* __restrict__ wsA) {
    int idx = blockIdx.x * blockDim.x + threadIdx.x;   // (mt, ch, lane)
    int total = MT_TOT * NCHUNK_P * 32;
    if (idx >= total) return;
    int lane = idx & 31;
    int ch   = (idx >> 5) % NCHUNK_P;
    int mt   = (idx >> 5) / NCHUNK_P;
    int hi   = lane >> 4;
    int n    = mt * 16 + (lane & 15);
    _Float16* dst = wsA + (size_t)idx * 16;
#pragma unroll
    for (int h = 0; h < 16; ++h) {
        int k = ch * 32 + 16 * hi + h;
        dst[h] = (k < K_TOT) ? (_Float16)w0[(size_t)k * SIZE_MEM + n]
                             : (_Float16)0.0f;
    }
}

// ---------------------------------------------------------------------------
// Prep: w_out (2048x64 fp32) -> f16 B-fragments for the output GEMM.
// B 32x16 layout: lane l half h = B[kbase + a_koff(h, l>=16)][nt*16 + l%16].
// ---------------------------------------------------------------------------
__global__ void k_prep_wout(const float* __restrict__ w_out,
                            _Float16* __restrict__ wsB) {
    int idx = blockIdx.x * blockDim.x + threadIdx.x;   // (nt, ch, lane)
    int total = (SIZE_OUT / 16) * OCHUNK * 32;
    if (idx >= total) return;
    int lane = idx & 31;
    int ch   = (idx >> 5) % OCHUNK;
    int nt   = (idx >> 5) / OCHUNK;
    int hi   = lane >> 4;
    int o    = nt * 16 + (lane & 15);
    _Float16* dst = wsB + (size_t)idx * 16;
#pragma unroll
    for (int h = 0; h < 16; ++h) {
        int k = ch * 32 + a_koff(h, hi);
        dst[h] = (_Float16)w_out[(size_t)k * SIZE_OUT + o];
    }
}

// ---------------------------------------------------------------------------
// Recurrent phase. G_WG blocks x 512 threads (16 waves).
// wave = 4*kq + mtile: 4 waves (kq=0..3) cooperate on one 16-neuron M-tile,
// each covering 17 K-chunks; partials reduced through LDS.
// GEMV via WMMA: B replicates z across all 16 columns -> D column 0
// (lanes 0 and 16) carries the result.
// ---------------------------------------------------------------------------
__global__ void __launch_bounds__(512) k_rnn_recur(
    const float* __restrict__ x, const float* __restrict__ b0,
    const _Float16* __restrict__ wsA, _Float16* __restrict__ mem_all,
    unsigned* __restrict__ bar) {
    __shared__ __align__(32) _Float16 z[K_PAD];            // 4352 B
    __shared__ __align__(32) float red[3 * MT_PER_WG * 32 * 8];  // 12 KB

    const int tid   = threadIdx.x;
    const int lane  = tid & 31;
    const int wave  = tid >> 5;        // 0..15
    const int mtile = wave & 3;        // M-tile within WG
    const int kq    = wave >> 2;       // K-quarter 0..3
    const int hi    = lane >> 4;
    const int mtg   = blockIdx.x * MT_PER_WG + mtile;      // global M-tile
    const size_t fragBase =
        ((size_t)mtg * NCHUNK_P + kq * CH_PER_W) * 32 * 16;

    // bias for the 8 neurons this lane can produce (D: VGPR j -> M = j+8*hi)
    float bias[8];
#pragma unroll
    for (int j = 0; j < 8; ++j)
        bias[j] = b0[mtg * 16 + j + 8 * hi];

    // zero the K padding tail once (never overwritten afterwards)
    if (tid < K_PAD - K_TOT) z[K_TOT + tid] = (_Float16)0.0f;

    const _Float16* fr = wsA + fragBase + (size_t)lane * 16;
    const _Float16* zq0 = z + kq * CH_PER_W * 32 + 16 * hi;

    for (int t = 0; t < T_STEPS; ++t) {
        // ---- stage z = [x_t, mem_{t-1}] into LDS as f16 ----
        if (tid < SIZE_IN)
            z[tid] = (_Float16)x[(size_t)t * SIZE_IN + tid];
        if (tid < 128) {
            if (t == 0) {
#pragma unroll
                for (int i = 0; i < 16; ++i)
                    z[SIZE_IN + tid * 16 + i] = (_Float16)0.0f;
            } else {
                const v16h* src = (const v16h*)(mem_all +
                    (size_t)(t - 1) * SIZE_MEM + tid * 16);
                *(v16h*)(z + SIZE_IN + tid * 16) = *src;
            }
        }
        // prefetch next step's x row into cache (global_prefetch_b8)
        if (tid == 0 && t + 1 < T_STEPS)
            __builtin_prefetch(x + (size_t)(t + 1) * SIZE_IN, 0, 1);
        __syncthreads();

        // ---- 17 chained WMMA over this wave's K quarter ----
        // B loads software-pipelined at depth 2 so ds_load_b128 for chunk
        // i+2 is in flight while wmma i executes.
        v8f c = {};
        v16h bA = *(const v16h*)(zq0);
        v16h bB = *(const v16h*)(zq0 + 32);
#pragma unroll
        for (int ch = 0; ch < CH_PER_W; ++ch) {
            v16h bC;
            if (ch + 2 < CH_PER_W)
                bC = *(const v16h*)(zq0 + (ch + 2) * 32);
            v16h a = *(const v16h*)(fr + (size_t)ch * 32 * 16);
            c = __builtin_amdgcn_wmma_f32_16x16x32_f16(
                    false, a, false, bA, (short)0, c, false, false);
            bA = bB;
            bB = bC;
        }

        // ---- reduce 4 K-partials through LDS ----
        if (kq != 0)
            *(v8f*)(red + (((kq - 1) * MT_PER_WG + mtile) * 32 + lane) * 8) = c;
        __syncthreads();
        if (kq == 0) {
#pragma unroll
            for (int p = 0; p < 3; ++p) {
                v8f r = *(const v8f*)(red +
                    ((p * MT_PER_WG + mtile) * 32 + lane) * 8);
#pragma unroll
                for (int j = 0; j < 8; ++j) c[j] += r[j];
            }
            // column-0 lanes hold the GEMV result; 8 contiguous neurons each
            if ((lane & 15) == 0) {
                v8h pk;
#pragma unroll
                for (int j = 0; j < 8; ++j)
                    pk[j] = (_Float16)fast_sigmoid(c[j] + bias[j]);
                *(v8h*)(mem_all + (size_t)t * SIZE_MEM + mtg * 16 + 8 * hi) = pk;
            }
        }

        // ---- grid barrier: monotonic arrival counter ----
        __syncthreads();
        if (tid == 0) {
            __threadfence();
            atomicAdd(bar, 1u);
            const unsigned target = (unsigned)(G_WG * (t + 1));
            while (__hip_atomic_load(bar, __ATOMIC_RELAXED,
                                     __HIP_MEMORY_SCOPE_AGENT) < target)
                __builtin_amdgcn_s_sleep(1);
        }
        __syncthreads();
    }
}

// ---------------------------------------------------------------------------
// Output GEMM: out = sigmoid(mem_all @ w_out + b_out), 8192x64x2048, dense.
// 2048 wave-tiles (512 M-tiles x 4 N-tiles) = 256 blocks x 8 waves.
// ---------------------------------------------------------------------------
__global__ void __launch_bounds__(256) k_out_gemm(
    const _Float16* __restrict__ mem_all, const _Float16* __restrict__ wsB,
    const float* __restrict__ b_out, float* __restrict__ out) {
    const int lane = threadIdx.x & 31;
    const int wid  = (blockIdx.x * blockDim.x + threadIdx.x) >> 5;
    const int mt   = wid >> 2;           // 0..511
    const int nt   = wid & 3;            // 0..3
    const int hi   = lane >> 4;
    const int row  = mt * 16 + (lane & 15);

    v8f c = {};
    const _Float16* bbase = wsB + ((size_t)nt * OCHUNK * 32 + lane) * 16;
    const _Float16* arow  = mem_all + (size_t)row * SIZE_MEM + 8 * hi;
#pragma unroll 2
    for (int ch = 0; ch < OCHUNK; ++ch) {
        v8h a0 = *(const v8h*)(arow + ch * 32);
        v8h a1 = *(const v8h*)(arow + ch * 32 + 16);
        v16h a;
#pragma unroll
        for (int i = 0; i < 8; ++i) { a[i] = a0[i]; a[8 + i] = a1[i]; }
        v16h b = *(const v16h*)(bbase + (size_t)ch * 32 * 16);
        c = __builtin_amdgcn_wmma_f32_16x16x32_f16(
                false, a, false, b, (short)0, c, false, false);
    }

    const int o = nt * 16 + (lane & 15);
    const float bias = b_out[o];
#pragma unroll
    for (int j = 0; j < 8; ++j) {
        int t = mt * 16 + j + 8 * hi;
        out[(size_t)t * SIZE_OUT + o] = fast_sigmoid(c[j] + bias);
    }
}

// ---------------------------------------------------------------------------
extern "C" void kernel_launch(void* const* d_in, const int* in_sizes, int n_in,
                              void* d_out, int out_size, void* d_ws,
                              size_t ws_size, hipStream_t stream) {
    (void)in_sizes; (void)n_in; (void)out_size; (void)ws_size;
    const float* x    = (const float*)d_in[0];   // [8192, 64]
    const float* w0   = (const float*)d_in[1];   // [2112, 2048]
    const float* b0   = (const float*)d_in[2];   // [1, 2048]
    const float* wout = (const float*)d_in[3];   // [2048, 64]
    const float* bout = (const float*)d_in[4];   // [1, 64]
    float* out = (float*)d_out;                  // [8192, 64]

    char* ws = (char*)d_ws;
    unsigned* bar     = (unsigned*)ws;                                  // 256 B
    _Float16* mem_all = (_Float16*)(ws + 256);                          // 32 MB
    _Float16* wsA     = mem_all + (size_t)T_STEPS * SIZE_MEM;           // 8.9 MB
    _Float16* wsB     = wsA + (size_t)MT_TOT * NCHUNK_P * 32 * 16;      // 256 KB

    hipMemsetAsync(bar, 0, 256, stream);

    const int totA = MT_TOT * NCHUNK_P * 32;          // 278528 frag groups
    k_prep_w0<<<(totA + 255) / 256, 256, 0, stream>>>(w0, wsA);
    const int totB = (SIZE_OUT / 16) * OCHUNK * 32;   // 8192 frag groups
    k_prep_wout<<<(totB + 255) / 256, 256, 0, stream>>>(wout, wsB);

    k_rnn_recur<<<G_WG, 512, 0, stream>>>(x, b0, wsA, mem_all, bar);
    k_out_gemm<<<256, 256, 0, stream>>>(mem_all, wsB, bout, out);
}
